// Compression_module_35373350650553
// MI455X (gfx1250) — compile-verified, matrix operand order checked by reading, not run on hardware
//
#include <hip/hip_runtime.h>
#include <hip/hip_bf16.h>
#include <stdint.h>

// Problem constants
#define B_    16
#define N_    1024
#define CIN   768
#define H_    8
#define KQ_   256
#define VD_   512
#define NHV   4096
#define NQKV  8192
#define DOUT  512
#define EPS_  1e-5f
#define SCALE_ 0.044194173824159216f   // 512^-0.5

typedef __attribute__((ext_vector_type(16))) __bf16 bf16x16;
typedef __attribute__((ext_vector_type(8)))  float  f32x8;

union FragU { bf16x16 v; uint4 q[2]; };

// Hardware packed f32x2 -> bf16x2 (V_CVT_PK_BF16_F32): D[15:0]=cvt(a), D[31:16]=cvt(b)
__device__ __forceinline__ unsigned pack2(float a, float b) {
  unsigned r;
  asm("v_cvt_pk_bf16_f32 %0, %1, %2" : "=v"(r) : "v"(a), "v"(b));
  return r;
}
__device__ __forceinline__ unsigned short f2bf_bits(float x) {
  return (unsigned short)pack2(x, 0.f);
}

// Async direct global->LDS copy, 16 bytes per lane (GLOBAL_LOAD_ASYNC_TO_LDS_B128,
// tracked by ASYNCcnt). Bypasses VGPRs entirely.
__device__ __forceinline__ void async_g2l_b128(const void* g, void* l) {
  unsigned lds = (unsigned)(size_t)l;                 // addr[31:0] = LDS offset
  unsigned long long ga = (unsigned long long)(size_t)g;
  asm volatile("global_load_async_to_lds_b128 %0, %1, off"
               :: "v"(lds), "v"(ga) : "memory");
}
__device__ __forceinline__ void wait_async0() {
#if __has_builtin(__builtin_amdgcn_s_wait_asynccnt)
  __builtin_amdgcn_s_wait_asynccnt(0);
#else
  asm volatile("s_wait_asynccnt 0" ::: "memory");
#endif
}
// Scheduling fence: keep prefetch loads issued before the WMMA block
// (prevents the backend from sinking them to their use site).
__device__ __forceinline__ void sched_fence() {
#if __has_builtin(__builtin_amdgcn_sched_barrier)
  __builtin_amdgcn_sched_barrier(0);
#endif
}

// Load one 16x32 bf16 WMMA fragment (A-row-major or B-col-major, identical
// lane layout for NT GEMM) from LDS. Per ISA 7.12.2: lanes 0-15 hold row
// (lane&15) with K = {0..7, 16..23}; lanes 16-31 hold K = {8..15, 24..31}.
__device__ __forceinline__ bf16x16 ld_frag(const __bf16* p, int ld) {
  int lane = threadIdx.x & 31;
  const __bf16* q = p + (lane & 15) * ld + ((lane >> 4) << 3);
  FragU f;
  f.q[0] = *(const uint4*)(q);        // K = kb .. kb+7
  f.q[1] = *(const uint4*)(q + 16);   // K = kb+16 .. kb+23
  return f.v;
}

// ---------------------------------------------------------------------------
// Kernel 1: qkv = BN(x @ Wqkv^T); scatter q,k row-major bf16 and v transposed.
// Tiles: 128(M) x 128(N) x 32(K); 8 waves as 2x4; per wave 64x32 (4x2 WMMA).
// Double-buffered LDS, software-pipelined: load(it+1) -> wmma(it) -> store(it+1).
// LDS rows padded to 40 elems (80B) -> conflict-free ds_load_b128.
// ---------------------------------------------------------------------------
#define LDG 40

__global__ __launch_bounds__(256)
void qkv_gemm_bn(const float* __restrict__ X, const float* __restrict__ W,
                 const float* __restrict__ gamma, const float* __restrict__ beta,
                 const float* __restrict__ mean,  const float* __restrict__ var,
                 unsigned short* __restrict__ qb, unsigned short* __restrict__ kb,
                 unsigned short* __restrict__ vtb)
{
  __shared__ __bf16 As[2][128 * LDG];
  __shared__ __bf16 Bs[2][128 * LDG];
  const int tid  = threadIdx.x;
  const int lane = tid & 31, wave = tid >> 5;
  const int wm = wave >> 2, wn = wave & 3;
  const int m0 = blockIdx.y * 128;
  const int n0 = blockIdx.x * 128;

  f32x8 acc[4][2];
  for (int i = 0; i < 4; i++) for (int j = 0; j < 2; j++)
    for (int e = 0; e < 8; e++) acc[i][j][e] = 0.f;

  const int lr = tid >> 1;          // 0..127: tile row this thread loads
  const int lh = (tid & 1) * 16;    // 0 or 16: k-half
  const float* asrc = X + (size_t)(m0 + lr) * CIN + lh;
  const float* bsrc = W + (size_t)(n0 + lr) * CIN + lh;

  float4 ra0, ra1, ra2, ra3, rb0, rb1, rb2, rb3;
  auto load_regs = [&](int kt) {
    const float* a = asrc + kt;
    ra0 = *(const float4*)(a + 0);  ra1 = *(const float4*)(a + 4);
    ra2 = *(const float4*)(a + 8);  ra3 = *(const float4*)(a + 12);
    const float* bp = bsrc + kt;
    rb0 = *(const float4*)(bp + 0); rb1 = *(const float4*)(bp + 4);
    rb2 = *(const float4*)(bp + 8); rb3 = *(const float4*)(bp + 12);
  };
  auto store_regs = [&](int buf) {
    __bf16* Ad = &As[buf][lr * LDG + lh];
    *(uint4*)(Ad)     = make_uint4(pack2(ra0.x,ra0.y), pack2(ra0.z,ra0.w), pack2(ra1.x,ra1.y), pack2(ra1.z,ra1.w));
    *(uint4*)(Ad + 8) = make_uint4(pack2(ra2.x,ra2.y), pack2(ra2.z,ra2.w), pack2(ra3.x,ra3.y), pack2(ra3.z,ra3.w));
    __bf16* Bd = &Bs[buf][lr * LDG + lh];
    *(uint4*)(Bd)     = make_uint4(pack2(rb0.x,rb0.y), pack2(rb0.z,rb0.w), pack2(rb1.x,rb1.y), pack2(rb1.z,rb1.w));
    *(uint4*)(Bd + 8) = make_uint4(pack2(rb2.x,rb2.y), pack2(rb2.z,rb2.w), pack2(rb3.x,rb3.y), pack2(rb3.z,rb3.w));
  };

  load_regs(0);
  store_regs(0);
  __syncthreads();

  const int NIT = CIN / 32;
  for (int it = 0; it < NIT; ++it) {
    const int cur = it & 1;
    if (it + 1 < NIT) load_regs((it + 1) * 32);   // global loads in flight
    sched_fence();                                 // ...and keep them there
    bf16x16 af[4], bfr[2];
    for (int i = 0; i < 4; i++) af[i]  = ld_frag(&As[cur][(wm * 64 + i * 16) * LDG], LDG);
    for (int j = 0; j < 2; j++) bfr[j] = ld_frag(&Bs[cur][(wn * 32 + j * 16) * LDG], LDG);
    for (int i = 0; i < 4; i++)
      for (int j = 0; j < 2; j++)
        acc[i][j] = __builtin_amdgcn_wmma_f32_16x16x32_bf16(
            false, af[i], false, bfr[j], (short)0, acc[i][j], false, false);
    if (it + 1 < NIT) store_regs(cur ^ 1);        // convert + stage next buffer
    __syncthreads();
  }

  // Fused BN epilogue + scatter to q / k / v^T (bf16).
  const int hh = (lane >> 4) * 8;     // row offset inside 16x16 tile
  const int ln = lane & 15;           // column inside tile
  for (int j = 0; j < 2; j++) {
    const int c = n0 + wn * 32 + j * 16 + ln;    // output channel
    const float sc = gamma[c] * rsqrtf(var[c] + EPS_);
    const float bi = beta[c] - mean[c] * sc;
    const int h = c >> 10;                       // head
    const int r = c & 1023;                      // channel within head
    for (int i = 0; i < 4; i++) {
      const int mrow0 = m0 + wm * 64 + i * 16 + hh;
      for (int e = 0; e < 8; e++) {
        const int mrow = mrow0 + e;
        const int bidx = mrow >> 10;             // batch
        const int nidx = mrow & 1023;            // token
        const unsigned short yb = f2bf_bits(acc[i][j][e] * sc + bi);
        if (r < KQ_) {
          qb[(((size_t)bidx * H_ + h) * N_ + nidx) * KQ_ + r] = yb;
        } else if (r < 2 * KQ_) {
          kb[(((size_t)bidx * H_ + h) * N_ + nidx) * KQ_ + (r - KQ_)] = yb;
        } else {
          vtb[(((size_t)bidx * H_ + h) * VD_ + (r - 2 * KQ_)) * N_ + nidx] = yb;
        }
      }
    }
  }
}

// ---------------------------------------------------------------------------
// Kernel 2: flash attention. One workgroup = (b, h, 64-row q tile).
// K/V tiles double-buffered and staged with GLOBAL_LOAD_ASYNC_TO_LDS_B128:
// next m-tile's async loads are issued right after the top barrier and only
// waited at the next iteration boundary, overlapping S/softmax/O compute.
// ---------------------------------------------------------------------------
#define LDQ 264   // 256 + 8 pad  -> conflict-free b128 frag loads
#define LDV 72    // 64 + 8 pad
#define SM_QS   0
#define SM_KS0  (SM_QS  + 64 * LDQ * 2)
#define SM_KS1  (SM_KS0 + 64 * LDQ * 2)
#define SM_VS0  (SM_KS1 + 64 * LDQ * 2)
#define SM_VS1  (SM_VS0 + 512 * LDV * 2)
#define SM_PS   (SM_VS1 + 512 * LDV * 2)
#define SM_RMAX (SM_PS + 64 * LDV * 2)
#define SM_RSUM (SM_RMAX + 4 * 64 * 4)
#define SM_M    (SM_RSUM + 4 * 64 * 4)
#define SM_L    (SM_M + 64 * 4)
#define SM_SCL  (SM_L + 64 * 4)
#define SM_TOTAL (SM_SCL + 64 * 4)

__global__ __launch_bounds__(256)
void flash_attn(const unsigned short* __restrict__ qb,
                const unsigned short* __restrict__ kb,
                const unsigned short* __restrict__ vtb,
                const float* __restrict__ pbias,
                unsigned short* __restrict__ out)
{
  extern __shared__ char smem[];
  __bf16* Qs  = (__bf16*)(smem + SM_QS);
  __bf16* Ks0 = (__bf16*)(smem + SM_KS0);
  __bf16* Ks1 = (__bf16*)(smem + SM_KS1);
  __bf16* Vs0 = (__bf16*)(smem + SM_VS0);
  __bf16* Vs1 = (__bf16*)(smem + SM_VS1);
  __bf16* Ps  = (__bf16*)(smem + SM_PS);
  float* redmax    = (float*)(smem + SM_RMAX);  // [4][64]
  float* redsum    = (float*)(smem + SM_RSUM);  // [4][64]
  float* rowmax_s  = (float*)(smem + SM_M);
  float* rowsum_s  = (float*)(smem + SM_L);
  float* rowscale_s= (float*)(smem + SM_SCL);

  const int tid  = threadIdx.x;
  const int lane = tid & 31, wave = tid >> 5;
  const int n0 = blockIdx.x * 64;
  const int h  = blockIdx.y;
  const int b  = blockIdx.z;
  const int hh = (lane >> 4) * 8;
  const int ln = lane & 15;

  const unsigned short* qbase = qb  + (((size_t)b * H_ + h) * N_ + n0) * KQ_;
  const unsigned short* kbase = kb  + ((size_t)b * H_ + h) * (size_t)N_ * KQ_;
  const unsigned short* vbase = vtb + ((size_t)b * H_ + h) * (size_t)VD_ * N_;
  const float*          pbase = pbias + ((size_t)h * N_ + n0) * N_;

  const int qrow = tid >> 2, qseg = (tid & 3) * 64;  // 4 threads/row, 128B each

  auto stage_kv = [&](int m0, int buf) {
    __bf16* Kd = buf ? Ks1 : Ks0;
    __bf16* Vd = buf ? Vs1 : Vs0;
    for (int u = 0; u < 8; u++)
      async_g2l_b128(kbase + (size_t)(m0 + qrow) * KQ_ + qseg + u * 8,
                     &Kd[qrow * LDQ + qseg + u * 8]);
    const int d0 = tid * 2;
    for (int rr = 0; rr < 2; rr++)
      for (int u = 0; u < 4; u++)
        async_g2l_b128(vbase + (size_t)(d0 + rr) * N_ + m0 + u * 8,
                       &Vd[(d0 + rr) * LDV + u * 8]);
  };

  { // async-load Q tile 64x256 once (covered by first boundary wait)
    for (int u = 0; u < 8; u++)
      async_g2l_b128(qbase + (size_t)qrow * KQ_ + qseg + u * 8,
                     &Qs[qrow * LDQ + qseg + u * 8]);
  }
  stage_kv(0, 0);
  if (tid < 64) { rowmax_s[tid] = -1e30f; rowsum_s[tid] = 0.f; }

  f32x8 oacc[4][4];
  for (int i = 0; i < 4; i++) for (int j = 0; j < 4; j++)
    for (int e = 0; e < 8; e++) oacc[i][j][e] = 0.f;

  const int NMT = N_ / 64;
  for (int it = 0; it < NMT; ++it) {
    const int cur = it & 1;
    const int m0 = it * 64;
    wait_async0();       // tiles for this iteration landed in LDS
    __syncthreads();
    if (it + 1 < NMT) stage_kv(m0 + 64, cur ^ 1);   // overlap with compute
    const __bf16* Ksc = cur ? Ks1 : Ks0;
    const __bf16* Vsc = cur ? Vs1 : Vs0;

    // S tiles: wave handles tiles {wave, wave+8} of the 4x4 grid
    f32x8 st[2];
    int tr[2], tc[2];
    for (int t = 0; t < 2; t++) {
      const int tile = wave + t * 8;
      const int r = tile >> 2, c = tile & 3;
      tr[t] = r; tc[t] = c;
      f32x8 a; for (int e = 0; e < 8; e++) a[e] = 0.f;
      for (int kc = 0; kc < 8; kc++) {
        bf16x16 qa = ld_frag(&Qs[(r * 16) * LDQ + kc * 32], LDQ);
        bf16x16 kf = ld_frag(&Ksc[(c * 16) * LDQ + kc * 32], LDQ);
        a = __builtin_amdgcn_wmma_f32_16x16x32_bf16(
            false, qa, false, kf, (short)0, a, false, false);
      }
      const float* pbp = pbase + (size_t)(r * 16 + hh) * N_ + m0 + c * 16 + ln;
      for (int e = 0; e < 8; e++)
        a[e] = a[e] * SCALE_ + pbp[(size_t)e * N_];
      // per-row max across the 16 lanes holding this row
      for (int e = 0; e < 8; e++) {
        float m = a[e];
        for (int mask = 1; mask < 16; mask <<= 1)
          m = fmaxf(m, __shfl_xor(m, mask, 32));
        if (ln == 0) redmax[c * 64 + r * 16 + hh + e] = m;
      }
      st[t] = a;
    }
    __syncthreads();
    if (tid < 64) {  // merge tile maxima, update running stats
      const float tm = fmaxf(fmaxf(redmax[tid], redmax[64 + tid]),
                             fmaxf(redmax[128 + tid], redmax[192 + tid]));
      const float mold = rowmax_s[tid];
      const float mnew = fmaxf(mold, tm);
      const float scl  = __expf(mold - mnew);
      rowscale_s[tid] = scl;
      rowsum_s[tid]  *= scl;
      rowmax_s[tid]   = mnew;
    }
    __syncthreads();
    // P = exp(S - mnew) -> LDS (bf16); partial row sums
    for (int t = 0; t < 2; t++) {
      const int r = tr[t], c = tc[t];
      for (int e = 0; e < 8; e++) {
        const int row = r * 16 + hh + e;
        const float p = __expf(st[t][e] - rowmax_s[row]);
        float su = p;
        for (int mask = 1; mask < 16; mask <<= 1)
          su += __shfl_xor(su, mask, 32);
        if (ln == 0) redsum[c * 64 + row] = su;
        ((unsigned short*)Ps)[row * LDV + c * 16 + ln] = f2bf_bits(p);
      }
    }
    // rescale running O by exp(mold - mnew)
    for (int i = 0; i < 4; i++)
      for (int e = 0; e < 8; e++) {
        const float scl = rowscale_s[i * 16 + hh + e];
        for (int j = 0; j < 4; j++) oacc[i][j][e] *= scl;
      }
    __syncthreads();
    if (tid < 64)
      rowsum_s[tid] += redsum[tid] + redsum[64 + tid] + redsum[128 + tid] + redsum[192 + tid];
    // O += P @ V ; this wave owns v-dims [wave*64, wave*64+64)
    for (int kc = 0; kc < 2; kc++) {
      bf16x16 pf[4];
      for (int i = 0; i < 4; i++) pf[i] = ld_frag(&Ps[(i * 16) * LDV + kc * 32], LDV);
      for (int j = 0; j < 4; j++) {
        bf16x16 vf = ld_frag(&Vsc[(wave * 64 + j * 16) * LDV + kc * 32], LDV);
        for (int i = 0; i < 4; i++)
          oacc[i][j] = __builtin_amdgcn_wmma_f32_16x16x32_bf16(
              false, pf[i], false, vf, (short)0, oacc[i][j], false, false);
      }
    }
  }
  __syncthreads();

  // epilogue: normalize, hardtanh, bf16 store to [B, N, H*VD]
  for (int i = 0; i < 4; i++) {
    for (int e = 0; e < 8; e++) {
      const int row = i * 16 + hh + e;
      const float inv = 1.f / rowsum_s[row];
      const size_t base = ((size_t)b * N_ + n0 + row) * NHV + (size_t)h * VD_;
      for (int j = 0; j < 4; j++) {
        float v = oacc[i][j][e] * inv;
        v = fminf(1.f, fmaxf(-1.f, v));
        out[base + wave * 64 + j * 16 + ln] = f2bf_bits(v);
      }
    }
  }
}

// ---------------------------------------------------------------------------
// Kernel 3: out = BN(attn_out @ Wproj^T); A bf16 staged via async-to-LDS,
// B converted f32->bf16 through VGPRs; double-buffered + pipelined; f32 out.
// ---------------------------------------------------------------------------
__global__ __launch_bounds__(256)
void proj_gemm_bn(const unsigned short* __restrict__ A, const float* __restrict__ W,
                  const float* __restrict__ gamma, const float* __restrict__ beta,
                  const float* __restrict__ mean,  const float* __restrict__ var,
                  float* __restrict__ out)
{
  __shared__ __bf16 As[2][128 * LDG];
  __shared__ __bf16 Bs[2][128 * LDG];
  const int tid  = threadIdx.x;
  const int lane = tid & 31, wave = tid >> 5;
  const int wm = wave >> 2, wn = wave & 3;
  const int m0 = blockIdx.y * 128;
  const int n0 = blockIdx.x * 128;

  f32x8 acc[4][2];
  for (int i = 0; i < 4; i++) for (int j = 0; j < 2; j++)
    for (int e = 0; e < 8; e++) acc[i][j][e] = 0.f;

  const int lr = tid >> 1;
  const int lh = (tid & 1) * 16;
  const unsigned short* asrc = A + (size_t)(m0 + lr) * NHV + lh;
  const float*          bsrc = W + (size_t)(n0 + lr) * NHV + lh;

  float4 rb0, rb1, rb2, rb3;
  auto stage_a = [&](int kt, int buf) {   // direct async global->LDS (bf16)
    async_g2l_b128(asrc + kt,     &As[buf][lr * LDG + lh]);
    async_g2l_b128(asrc + kt + 8, &As[buf][lr * LDG + lh + 8]);
  };
  auto load_b = [&](int kt) {
    const float* bp = bsrc + kt;
    rb0 = *(const float4*)(bp + 0); rb1 = *(const float4*)(bp + 4);
    rb2 = *(const float4*)(bp + 8); rb3 = *(const float4*)(bp + 12);
  };
  auto store_b = [&](int buf) {
    __bf16* Bd = &Bs[buf][lr * LDG + lh];
    *(uint4*)(Bd)     = make_uint4(pack2(rb0.x,rb0.y), pack2(rb0.z,rb0.w), pack2(rb1.x,rb1.y), pack2(rb1.z,rb1.w));
    *(uint4*)(Bd + 8) = make_uint4(pack2(rb2.x,rb2.y), pack2(rb2.z,rb2.w), pack2(rb3.x,rb3.y), pack2(rb3.z,rb3.w));
  };

  stage_a(0, 0);
  load_b(0);
  store_b(0);
  wait_async0();
  __syncthreads();

  const int NIT = NHV / 32;
  for (int it = 0; it < NIT; ++it) {
    const int cur = it & 1;
    if (it + 1 < NIT) {
      stage_a((it + 1) * 32, cur ^ 1);   // async fills next buffer during wmma
      load_b((it + 1) * 32);             // global loads in flight
    }
    sched_fence();                        // keep the loads issued up here
    bf16x16 af[4], bfr[2];
    for (int i = 0; i < 4; i++) af[i]  = ld_frag(&As[cur][(wm * 64 + i * 16) * LDG], LDG);
    for (int j = 0; j < 2; j++) bfr[j] = ld_frag(&Bs[cur][(wn * 32 + j * 16) * LDG], LDG);
    for (int i = 0; i < 4; i++)
      for (int j = 0; j < 2; j++)
        acc[i][j] = __builtin_amdgcn_wmma_f32_16x16x32_bf16(
            false, af[i], false, bfr[j], (short)0, acc[i][j], false, false);
    if (it + 1 < NIT) store_b(cur ^ 1);
    wait_async0();
    __syncthreads();
  }

  const int hh = (lane >> 4) * 8;
  const int ln = lane & 15;
  for (int j = 0; j < 2; j++) {
    const int c = n0 + wn * 32 + j * 16 + ln;
    const float sc = gamma[c] * rsqrtf(var[c] + EPS_);
    const float bi = beta[c] - mean[c] * sc;
    for (int i = 0; i < 4; i++) {
      const int mrow0 = m0 + wm * 64 + i * 16 + hh;
      for (int e = 0; e < 8; e++)
        out[(size_t)(mrow0 + e) * DOUT + c] = acc[i][j][e] * sc + bi;
    }
  }
}

// ---------------------------------------------------------------------------
extern "C" void kernel_launch(void* const* d_in, const int* in_sizes, int n_in,
                              void* d_out, int out_size, void* d_ws, size_t ws_size,
                              hipStream_t stream)
{
  (void)in_sizes; (void)n_in; (void)out_size; (void)ws_size;
  const float* x          = (const float*)d_in[0];
  const float* qkv_w      = (const float*)d_in[1];
  const float* qkv_gamma  = (const float*)d_in[2];
  const float* qkv_beta   = (const float*)d_in[3];
  const float* qkv_mean   = (const float*)d_in[4];
  const float* qkv_var    = (const float*)d_in[5];
  const float* pos_bias   = (const float*)d_in[6];
  const float* proj_w     = (const float*)d_in[7];
  const float* proj_gamma = (const float*)d_in[8];
  const float* proj_beta  = (const float*)d_in[9];
  const float* proj_mean  = (const float*)d_in[10];
  const float* proj_var   = (const float*)d_in[11];

  char* ws = (char*)d_ws;
  unsigned short* qb  = (unsigned short*)(ws);                        // 64 MiB
  unsigned short* kb  = (unsigned short*)(ws + ((size_t)64  << 20));  // 64 MiB
  unsigned short* vtb = (unsigned short*)(ws + ((size_t)128 << 20));  // 128 MiB
  unsigned short* ao  = (unsigned short*)(ws + ((size_t)256 << 20));  // 128 MiB

  dim3 blk(256);
  qkv_gemm_bn<<<dim3(NQKV / 128, (B_ * N_) / 128), blk, 0, stream>>>(
      x, qkv_w, qkv_gamma, qkv_beta, qkv_mean, qkv_var, qb, kb, vtb);
  flash_attn<<<dim3(N_ / 64, H_, B_), blk, SM_TOTAL, stream>>>(
      qb, kb, vtb, pos_bias, ao);
  proj_gemm_bn<<<dim3(DOUT / 128, (B_ * N_) / 128), blk, 0, stream>>>(
      ao, proj_w, proj_gamma, proj_beta, proj_mean, proj_var, (float*)d_out);
}
